// SelfAttentionKNN_81973745811802
// MI455X (gfx1250) — matrix operand order repaired and checked by reading, not compiled, not run on hardware
//
#include <hip/hip_runtime.h>
#include <hip/hip_bf16.h>
#include <stdint.h>

// ---------------------------------------------------------------------------
// Problem constants (fixed by reference setup_inputs)
// ---------------------------------------------------------------------------
constexpr int BB    = 2;
constexpr int NPT   = 8192;
constexpr int DD    = 256;          // input dim (= GEMM K for q/kv)
constexpr int HH    = 8;
constexpr int DH    = 64;
constexpr int INNER = HH * DH;      // 512
constexpr int KNN   = 8;
constexpr int BN    = BB * NPT;     // 16384 rows

typedef __attribute__((ext_vector_type(16))) __bf16        v16bf;
typedef __attribute__((ext_vector_type(8)))  float         v8f;
typedef __attribute__((ext_vector_type(8)))  unsigned int  v8u;

// ---------------------------------------------------------------------------
// bf16 helpers (round-to-nearest-even)
// ---------------------------------------------------------------------------
static __device__ __forceinline__ unsigned short f32_to_bf16(float f) {
  unsigned int u = __builtin_bit_cast(unsigned int, f);
  u += 0x7FFFu + ((u >> 16) & 1u);
  return (unsigned short)(u >> 16);
}
static __device__ __forceinline__ float bf16_to_f32(unsigned short h) {
  unsigned int u = ((unsigned int)h) << 16;
  return __builtin_bit_cast(float, u);
}

// ---------------------------------------------------------------------------
// Elementwise f32 -> bf16
// ---------------------------------------------------------------------------
__global__ void cvt_bf16_kernel(const float* __restrict__ in,
                                unsigned short* __restrict__ out, int n) {
  int i = blockIdx.x * blockDim.x + threadIdx.x;
  if (i < n) out[i] = f32_to_bf16(in[i]);
}

// in [rows, cols] f32 row-major  ->  out [cols, rows] bf16 row-major (B^T)
__global__ void transpose_bf16_kernel(const float* __restrict__ in,
                                      unsigned short* __restrict__ out,
                                      int rows, int cols) {
  int i = blockIdx.x * blockDim.x + threadIdx.x;
  if (i < rows * cols) {
    int r = i / cols, c = i % cols;
    out[(size_t)c * rows + r] = f32_to_bf16(in[i]);
  }
}

// ---------------------------------------------------------------------------
// KNN: top-8 smallest squared distances per query point.
// One thread per query; candidates tiled through LDS.
// ---------------------------------------------------------------------------
__global__ __launch_bounds__(256) void knn_kernel(const float* __restrict__ pos,
                                                  int* __restrict__ knn_idx) {
  const int b = blockIdx.y;
  const int q = blockIdx.x * 256 + threadIdx.x;
  const float* P = pos + (size_t)b * NPT * 3;

  const float qx = P[q * 3 + 0];
  const float qy = P[q * 3 + 1];
  const float qz = P[q * 3 + 2];

  float bd[KNN];
  int   bi[KNN];
#pragma unroll
  for (int r = 0; r < KNN; ++r) { bd[r] = 3.4e38f; bi[r] = 0; }

  __shared__ float sx[256], sy[256], sz[256];

  for (int t0 = 0; t0 < NPT; t0 += 256) {
    __syncthreads();
    {
      int c = t0 + threadIdx.x;
      sx[threadIdx.x] = P[c * 3 + 0];
      sy[threadIdx.x] = P[c * 3 + 1];
      sz[threadIdx.x] = P[c * 3 + 2];
    }
    __syncthreads();
    for (int j = 0; j < 256; ++j) {
      float dx = sx[j] - qx, dy = sy[j] - qy, dz = sz[j] - qz;
      float d2 = dx * dx + dy * dy + dz * dz;
      if (d2 < bd[KNN - 1]) {          // rare: sorted swap-down insert
        float dc = d2; int ic = t0 + j;
#pragma unroll
        for (int r = 0; r < KNN; ++r) {
          if (dc < bd[r]) {            // strict < keeps earlier index on ties
            float td = bd[r]; int ti = bi[r];
            bd[r] = dc; bi[r] = ic;
            dc = td;    ic = ti;
          }
        }
      }
    }
  }
  int* out = knn_idx + ((size_t)b * NPT + q) * KNN;
#pragma unroll
  for (int r = 0; r < KNN; ++r) out[r] = bi[r];
}

// ---------------------------------------------------------------------------
// WMMA bf16 GEMM:  C[M,Nc] = A[M,Kd] * B[Kd,Nc]  (B given transposed: Bt[Nc,Kd])
// One wave computes a 16x64 tile: A-fragment reused across 4 B-fragments.
// A/B fragments use contiguous k order K = 16*half + s, same for both (wave32).
// ---------------------------------------------------------------------------
static __device__ __forceinline__ v16bf load_frag_a(const unsigned short* p, int half) {
  // lane half 0: K = 0..15 ; half 1: K = 16..31 (contiguous 32B)
  uint4 lo = *reinterpret_cast<const uint4*>(p + half * 16);
  uint4 hi = *reinterpret_cast<const uint4*>(p + half * 16 + 8);
  v8u u = {lo.x, lo.y, lo.z, lo.w, hi.x, hi.y, hi.z, hi.w};
  return __builtin_bit_cast(v16bf, u);
}
static __device__ __forceinline__ v16bf load_frag_b(const unsigned short* p, int half) {
  // lane half 0: K = 0..15 ; half 1: K = 16..31 (contiguous 32B)
  const uint4* q = reinterpret_cast<const uint4*>(p + half * 16);
  uint4 lo = q[0], hi = q[1];
  v8u u = {lo.x, lo.y, lo.z, lo.w, hi.x, hi.y, hi.z, hi.w};
  return __builtin_bit_cast(v16bf, u);
}

template <int OUT_BF16>
__global__ __launch_bounds__(256) void gemm_bf16_wmma(
    const unsigned short* __restrict__ A,   // [M, Kd] bf16
    const unsigned short* __restrict__ Bt,  // [Nc, Kd] bf16 (B transposed)
    void* __restrict__ C,                   // bf16 [M,Nc] or f32 [M,Nc]
    const float* __restrict__ bias,         // f32 [Nc] or nullptr
    int M, int Nc, int Kd) {
  const int wid    = blockIdx.x * (blockDim.x >> 5) + (threadIdx.x >> 5);
  const int tilesN = Nc >> 6;
  const int tm     = wid / tilesN;
  const int tn     = wid % tilesN;
  if (tm * 16 >= M) return;                  // wave-uniform; survivors have EXEC all-1s

  const int lane = threadIdx.x & 31;
  const int l16  = lane & 15;
  const int half = lane >> 4;

  const unsigned short* arow  = A  + (size_t)(tm * 16 + l16) * Kd;
  const unsigned short* brow0 = Bt + (size_t)(tn * 64 +  0 + l16) * Kd;
  const unsigned short* brow1 = Bt + (size_t)(tn * 64 + 16 + l16) * Kd;
  const unsigned short* brow2 = Bt + (size_t)(tn * 64 + 32 + l16) * Kd;
  const unsigned short* brow3 = Bt + (size_t)(tn * 64 + 48 + l16) * Kd;

  v8f acc0 = {}, acc1 = {}, acc2 = {}, acc3 = {};
#pragma unroll 4
  for (int k0 = 0; k0 < Kd; k0 += 32) {
    v16bf af = load_frag_a(arow + k0, half);
    v16bf b0 = load_frag_b(brow0 + k0, half);
    v16bf b1 = load_frag_b(brow1 + k0, half);
    v16bf b2 = load_frag_b(brow2 + k0, half);
    v16bf b3 = load_frag_b(brow3 + k0, half);
    acc0 = __builtin_amdgcn_wmma_f32_16x16x32_bf16(false, af, false, b0, (short)0, acc0, false, false);
    acc1 = __builtin_amdgcn_wmma_f32_16x16x32_bf16(false, af, false, b1, (short)0, acc1, false, false);
    acc2 = __builtin_amdgcn_wmma_f32_16x16x32_bf16(false, af, false, b2, (short)0, acc2, false, false);
    acc3 = __builtin_amdgcn_wmma_f32_16x16x32_bf16(false, af, false, b3, (short)0, acc3, false, false);
  }

  // Epilogue. C layout: VGPR r of lane L holds (M = r + 8*half, N = l16) in tile.
#pragma unroll
  for (int nn = 0; nn < 4; ++nn) {
    v8f a = (nn == 0) ? acc0 : (nn == 1) ? acc1 : (nn == 2) ? acc2 : acc3;
    const int c  = tn * 64 + nn * 16 + l16;
    const float bs = (OUT_BF16 == 0 && bias != nullptr) ? bias[c] : 0.0f;
#pragma unroll
    for (int r = 0; r < 8; ++r) {
      const int m = tm * 16 + half * 8 + r;
      const float val = a[r] + bs;
      if (OUT_BF16)
        ((unsigned short*)C)[(size_t)m * Nc + c] = f32_to_bf16(val);
      else
        ((float*)C)[(size_t)m * Nc + c] = val;
    }
  }
}

// ---------------------------------------------------------------------------
// Attention over K=8 KNN neighbors. One wave per (point, head); each lane owns
// two DH elements. Dots reduced with __shfl_xor; softmax in registers.
// ---------------------------------------------------------------------------
__global__ __launch_bounds__(256) void attn_kernel(
    const unsigned short* __restrict__ qb,   // [BN, INNER] bf16
    const unsigned short* __restrict__ kvb,  // [BN, 2*INNER] bf16 (k | v)
    const int* __restrict__ knn_idx,         // [BN, KNN]
    unsigned short* __restrict__ attb) {     // [BN, INNER] bf16
  const int pt   = blockIdx.x;               // 0 .. BN-1
  const int h    = threadIdx.x >> 5;         // 0 .. 7
  const int lane = threadIdx.x & 31;
  const int rowbase = (pt / NPT) * NPT;      // batch offset for neighbor rows

  int nbr[KNN];
#pragma unroll
  for (int j = 0; j < KNN; ++j) nbr[j] = knn_idx[(size_t)pt * KNN + j];

  const unsigned int qp =
      *reinterpret_cast<const unsigned int*>(qb + (size_t)pt * INNER + h * DH + lane * 2);
  const float q0 = bf16_to_f32((unsigned short)(qp & 0xFFFF));
  const float q1 = bf16_to_f32((unsigned short)(qp >> 16));

  float dots[KNN], v0[KNN], v1[KNN];
#pragma unroll
  for (int j = 0; j < KNN; ++j) {
    const size_t base = (size_t)(rowbase + nbr[j]) * (2 * INNER) + h * DH + lane * 2;
    const unsigned int kp = *reinterpret_cast<const unsigned int*>(kvb + base);
    const unsigned int vp = *reinterpret_cast<const unsigned int*>(kvb + base + INNER);
    v0[j] = bf16_to_f32((unsigned short)(vp & 0xFFFF));
    v1[j] = bf16_to_f32((unsigned short)(vp >> 16));
    float part = q0 * bf16_to_f32((unsigned short)(kp & 0xFFFF)) +
                 q1 * bf16_to_f32((unsigned short)(kp >> 16));
#pragma unroll
    for (int o = 16; o >= 1; o >>= 1) part += __shfl_xor(part, o, 32);
    dots[j] = part * 0.125f;                 // DH^-0.5
  }

  float mx = dots[0];
#pragma unroll
  for (int j = 1; j < KNN; ++j) mx = fmaxf(mx, dots[j]);
  float e[KNN], s = 0.0f;
#pragma unroll
  for (int j = 0; j < KNN; ++j) { e[j] = __expf(dots[j] - mx); s += e[j]; }
  const float inv = 1.0f / s;

  float o0 = 0.0f, o1 = 0.0f;
#pragma unroll
  for (int j = 0; j < KNN; ++j) {
    const float a = e[j] * inv;
    o0 += a * v0[j];
    o1 += a * v1[j];
  }
  const unsigned int packed =
      (unsigned int)f32_to_bf16(o0) | ((unsigned int)f32_to_bf16(o1) << 16);
  *reinterpret_cast<unsigned int*>(attb + (size_t)pt * INNER + h * DH + lane * 2) = packed;
}

// ---------------------------------------------------------------------------
// Launcher
// ---------------------------------------------------------------------------
extern "C" void kernel_launch(void* const* d_in, const int* in_sizes, int n_in,
                              void* d_out, int out_size, void* d_ws, size_t ws_size,
                              hipStream_t stream) {
  const float* x    = (const float*)d_in[0];  // [B,N,D]
  const float* pos  = (const float*)d_in[1];  // [B,N,3]
  const float* Wq   = (const float*)d_in[2];  // [D, INNER]
  const float* Wkv  = (const float*)d_in[3];  // [D, 2*INNER]
  const float* Wout = (const float*)d_in[4];  // [INNER, D]
  const float* bout = (const float*)d_in[5];  // [D]
  float* out = (float*)d_out;                 // [B,N,D] f32

  // Workspace carve (256B aligned)
  uintptr_t p = (uintptr_t)d_ws;
  p = (p + 255) & ~(uintptr_t)255;
  auto carve = [&](size_t bytes) {
    void* r = (void*)p;
    p += (bytes + 255) & ~(size_t)255;
    return r;
  };
  int*            knn_idx = (int*)carve((size_t)BN * KNN * sizeof(int));
  unsigned short* xh      = (unsigned short*)carve((size_t)BN * DD * 2);
  unsigned short* WqT     = (unsigned short*)carve((size_t)INNER * DD * 2);
  unsigned short* WkvT    = (unsigned short*)carve((size_t)2 * INNER * DD * 2);
  unsigned short* WoutT   = (unsigned short*)carve((size_t)DD * INNER * 2);
  unsigned short* qbuf    = (unsigned short*)carve((size_t)BN * INNER * 2);
  unsigned short* kvbuf   = (unsigned short*)carve((size_t)BN * 2 * INNER * 2);
  unsigned short* attb    = (unsigned short*)carve((size_t)BN * INNER * 2);

  // 1) Precision conversion / weight transposition
  cvt_bf16_kernel<<<(BN * DD + 255) / 256, 256, 0, stream>>>(x, xh, BN * DD);
  transpose_bf16_kernel<<<(DD * INNER + 255) / 256, 256, 0, stream>>>(Wq, WqT, DD, INNER);
  transpose_bf16_kernel<<<(DD * 2 * INNER + 255) / 256, 256, 0, stream>>>(Wkv, WkvT, DD, 2 * INNER);
  transpose_bf16_kernel<<<(INNER * DD + 255) / 256, 256, 0, stream>>>(Wout, WoutT, INNER, DD);

  // 2) KNN (independent of GEMMs above)
  knn_kernel<<<dim3(NPT / 256, BB), 256, 0, stream>>>(pos, knn_idx);

  // 3) q = x @ Wq          (M=BN, Nc=INNER,   Kd=DD)  -> bf16
  {
    int tiles = (BN / 16) * (INNER / 64);
    gemm_bf16_wmma<1><<<tiles / 8, 256, 0, stream>>>(xh, WqT, (void*)qbuf, nullptr, BN, INNER, DD);
  }
  // 4) kv = x @ Wkv        (M=BN, Nc=2*INNER, Kd=DD)  -> bf16  (gather folded into attention)
  {
    int tiles = (BN / 16) * (2 * INNER / 64);
    gemm_bf16_wmma<1><<<tiles / 8, 256, 0, stream>>>(xh, WkvT, (void*)kvbuf, nullptr, BN, 2 * INNER, DD);
  }

  // 5) KNN attention -> att bf16
  attn_kernel<<<BN, 256, 0, stream>>>(qbuf, kvbuf, knn_idx, attb);

  // 6) out = att @ Wout + bout   (M=BN, Nc=DD, Kd=INNER) -> f32 into d_out
  {
    int tiles = (BN / 16) * (DD / 64);
    gemm_bf16_wmma<0><<<tiles / 8, 256, 0, stream>>>(attb, WoutT, (void*)out, bout, BN, DD, INNER);
  }
}
